// KgAdapterCrossAttention_71442486002189
// MI455X (gfx1250) — compile-verified, matrix operand order checked by reading, not run on hardware
//
#include <hip/hip_runtime.h>

#define HIDDEN 256
#define NHEADS 4
#define HDIM   64
#define SEQ    2048
#define BATCH  4
#define NEGF   (-3.4028234663852886e38f)

typedef _Float16 half_t;
typedef __attribute__((ext_vector_type(16))) _Float16     v16h;
typedef __attribute__((ext_vector_type(8)))  float        v8f;
typedef __attribute__((ext_vector_type(4)))  unsigned int v4u;
typedef __attribute__((ext_vector_type(8)))  int          v8i_;
typedef __attribute__((ext_vector_type(4)))  int          v4i_;

// 16-bit A-matrix 16x32 layout (ISA 7.12.2): VGPR v, lane-half hi
// holds K = base, base+1 packed in one dword.
__device__ __forceinline__ int a_kbase(int v, int hi) {
    return ((v & 4) << 2) + (hi ? 8 : 0) + ((v & 3) << 1);
}

__device__ __forceinline__ v8f wmma_f16(v16h a, v16h b, v8f c) {
    return __builtin_amdgcn_wmma_f32_16x16x32_f16(false, a, false, b,
                                                  (short)0, c, false, false);
}

__device__ __forceinline__ unsigned lds_byte_off(const void* p) {
    // LDS object generic address: high bits = shared aperture, [31:0] = LDS offset
    return (unsigned)(unsigned long long)(size_t)p;
}

// Issue a 2D f16 tile DMA: global (row-major, row stride `stride0` elements,
// row length tile_d0, tile_d1 rows) -> LDS contiguous. Tracked on TENSORcnt.
__device__ __forceinline__ void tdm_load_2d_f16(unsigned lds_off, const void* gaddr,
                                                unsigned tensor_d0, unsigned tensor_d1,
                                                unsigned tile_d0, unsigned tile_d1,
                                                unsigned stride0) {
    unsigned long long ga = (unsigned long long)(size_t)gaddr;
    v4u g0;
    g0[0] = 1u;                                    // count=1, user descriptor
    g0[1] = lds_off;                               // LDS byte address
    g0[2] = (unsigned)(ga & 0xffffffffu);          // global addr [31:0]
    g0[3] = (unsigned)((ga >> 32) & 0x1ffffffu)    // global addr [56:32]
          | (2u << 30);                            // type = 2 ("image")
    v8i_ g1;
    g1[0] = (int)(1u << 16);                       // data_size = 1 -> 2 bytes
    g1[1] = (int)((tensor_d0 & 0xffffu) << 16);    // tensor_dim0 [15:0]
    g1[2] = (int)(((tensor_d0 >> 16) & 0xffffu)    // tensor_dim0 [31:16]
          | ((tensor_d1 & 0xffffu) << 16));        // tensor_dim1 [15:0]
    g1[3] = (int)(((tensor_d1 >> 16) & 0xffffu)    // tensor_dim1 [31:16]
          | ((tile_d0 & 0xffffu) << 16));          // tile_dim0
    g1[4] = (int)(tile_d1 & 0xffffu);              // tile_dim1 (tile_dim2 = 0)
    g1[5] = (int)stride0;                          // tensor_dim0_stride [31:0]
    g1[6] = 0;                                     // stride0 [47:32], stride1 lo
    g1[7] = 0;
    v4i_ z4 = {0, 0, 0, 0};
    v8i_ z8 = {0, 0, 0, 0, 0, 0, 0, 0};
    // clang-23 / therock signature: 6 args
    __builtin_amdgcn_tensor_load_to_lds(g0, g1, z4, z4, z8, 0);
}

// y = x @ W^T, one wave computes a 16x16 f32 tile with 8 WMMAs over K=256.
// MODE 0: store f16 -> Q  [b][h][s][d]
// MODE 1: store f16 -> K  [b][h][s][d]
// MODE 2: store f16 -> Vt [b][h][d][s]   (transposed for PV B-matrix loads)
// MODE 3: x is f16, store f32 -> out [row][n]
template <int MODE>
__global__ void proj_kernel(const void* __restrict__ xv,
                            const float* __restrict__ W,
                            void* __restrict__ yv) {
    const int lane = threadIdx.x;
    const int wid  = blockIdx.x * 4 + threadIdx.y;
    const int NT   = HIDDEN / 16;          // 16 column tiles
    const int rt   = wid / NT;
    const int ct   = wid % NT;
    const int hi   = lane >> 4;
    const int lml  = lane & 15;
    const long rowBase = (long)rt * 16;

    v8f acc = {};
    for (int k0 = 0; k0 < HIDDEN; k0 += 32) {
        v16h a, b;
#pragma unroll
        for (int v = 0; v < 8; ++v) {
            const int kk = k0 + a_kbase(v, hi);
            if (MODE == 3) {
                const half_t* x = (const half_t*)xv;
                const half_t* xr = x + (rowBase + lml) * HIDDEN + kk;
                a[2 * v]     = xr[0];
                a[2 * v + 1] = xr[1];
            } else {
                const float* x = (const float*)xv;
                const float* xr = x + (rowBase + lml) * HIDDEN + kk;
                a[2 * v]     = (half_t)xr[0];
                a[2 * v + 1] = (half_t)xr[1];
            }
            const float* wr = W + (long)(ct * 16 + lml) * HIDDEN + kk;
            b[2 * v]     = (half_t)wr[0];
            b[2 * v + 1] = (half_t)wr[1];
        }
        acc = wmma_f16(a, b, acc);
    }

#pragma unroll
    for (int i = 0; i < 8; ++i) {
        const int  row = i + hi * 8;
        const long rm  = rowBase + row;
        const int  n   = ct * 16 + lml;
        const float val = acc[i];
        if (MODE == 3) {
            ((float*)yv)[rm * HIDDEN + n] = val;
        } else {
            const int bi = (int)(rm / SEQ), s = (int)(rm % SEQ);
            const int h  = n / HDIM, d = n % HDIM;
            half_t* y = (half_t*)yv;
            if (MODE == 2)
                y[((long)(bi * NHEADS + h) * HDIM + d) * SEQ + s] = (half_t)val;
            else
                y[((long)(bi * NHEADS + h) * SEQ + s) * HDIM + d] = (half_t)val;
        }
    }
}

// Fused flash-style attention. 4 waves/block share one (b,h); K/V tiles are
// staged into LDS by the Tensor Data Mover (double buffered, TENSORcnt),
// each wave owns a 16-query tile with online softmax in registers.
__global__ void attn_kernel(const half_t* __restrict__ Qb,
                            const half_t* __restrict__ Kb,
                            const half_t* __restrict__ Vt,
                            const float* __restrict__ amask,
                            const int* __restrict__ align,
                            half_t* __restrict__ Ob) {
    __shared__ half_t kTile[2][32][HDIM];   // [buf][key][d]      8 KB
    __shared__ half_t vTile[2][HDIM][32];   // [buf][d][key]      8 KB
    __shared__ half_t pbuf[4][16][34];      // P staging per wave

    const int lane = threadIdx.x;
    const int wave = threadIdx.y;
    const int hi   = lane >> 4;
    const int lml  = lane & 15;
    const int qt   = blockIdx.x * 4 + wave;
    const int h    = blockIdx.y;
    const int b    = blockIdx.z;
    const int qb0  = qt * 16;
    const long bh  = (long)b * NHEADS + h;

    const half_t* Qp = Qb + bh * SEQ * HDIM;
    const half_t* Kp = Kb + bh * SEQ * HDIM;
    const half_t* Vp = Vt + bh * HDIM * SEQ;

    // Preload Q A-tiles for the two d-chunks (d 0..31, 32..63).
    v16h aq0, aq1;
#pragma unroll
    for (int v = 0; v < 8; ++v) {
        const int kk = a_kbase(v, hi);
        const half_t* qr = Qp + (long)(qb0 + lml) * HDIM;
        aq0[2 * v] = qr[kk];          aq0[2 * v + 1] = qr[kk + 1];
        aq1[2 * v] = qr[32 + kk];     aq1[2 * v + 1] = qr[32 + kk + 1];
    }

    float mrow[8], lrow[8];
#pragma unroll
    for (int i = 0; i < 8; ++i) { mrow[i] = -INFINITY; lrow[i] = 0.0f; }
    v8f oc[4] = {};

    // Kick off DMA of the first K/V tile into buffer 0 (wave 0 only).
    if (wave == 0) {
        tdm_load_2d_f16(lds_byte_off(&kTile[0][0][0]), Kp,
                        HDIM, SEQ, HDIM, 32, HDIM);
        tdm_load_2d_f16(lds_byte_off(&vTile[0][0][0]), Vp,
                        SEQ, HDIM, 32, HDIM, SEQ);
    }

    for (int kb = 0, it = 0; kb < SEQ; kb += 32, ++it) {
        const int cur = it & 1;
        // Current tile landed -> make visible to all 4 waves.
        if (wave == 0) __builtin_amdgcn_s_wait_tensorcnt(0);
        __syncthreads();
        // Prefetch next tile into the other buffer (overlaps compute below).
        if (wave == 0 && kb + 32 < SEQ) {
            tdm_load_2d_f16(lds_byte_off(&kTile[cur ^ 1][0][0]),
                            Kp + (long)(kb + 32) * HDIM,
                            HDIM, SEQ, HDIM, 32, HDIM);
            tdm_load_2d_f16(lds_byte_off(&vTile[cur ^ 1][0][0]),
                            Vp + (kb + 32),
                            SEQ, HDIM, 32, HDIM, SEQ);
        }

        // ---- S = Q K^T for two 16-key tiles (B fragments from LDS) ----
        v8f s0 = {}, s1 = {};
#pragma unroll
        for (int c = 0; c < 2; ++c) {
            v16h bk0, bk1;
#pragma unroll
            for (int v = 0; v < 8; ++v) {
                const int kk = c * 32 + a_kbase(v, hi);
                bk0[2 * v] = kTile[cur][lml][kk];
                bk0[2 * v + 1] = kTile[cur][lml][kk + 1];
                bk1[2 * v] = kTile[cur][16 + lml][kk];
                bk1[2 * v + 1] = kTile[cur][16 + lml][kk + 1];
            }
            s0 = wmma_f16(c ? aq1 : aq0, bk0, s0);
            s1 = wmma_f16(c ? aq1 : aq0, bk1, s1);
        }

        // ---- scale + additive mask + align fill + clamp ----
        float s0v[8], s1v[8];
#pragma unroll
        for (int i = 0; i < 8; ++i) {
            const int row = i + 8 * hi;
            const int q   = qb0 + row;
            const int k0  = kb + lml;
            const int k1  = kb + 16 + lml;
            float x0 = s0[i] * 0.125f + amask[((long)b * SEQ + q) * SEQ + k0];
            float x1 = s1[i] * 0.125f + amask[((long)b * SEQ + q) * SEQ + k1];
            x0 = fmaxf(x0, NEGF);
            x1 = fmaxf(x1, NEGF);
            if (align[((long)b * SEQ + k0) * SEQ + q] == 0) x0 = NEGF;
            if (align[((long)b * SEQ + k1) * SEQ + q] == 0) x1 = NEGF;
            s0v[i] = x0;
            s1v[i] = x1;
        }

        // ---- online softmax update ----
#pragma unroll
        for (int i = 0; i < 8; ++i) {
            float rmax = fmaxf(s0v[i], s1v[i]);
#pragma unroll
            for (int m = 1; m < 16; m <<= 1)
                rmax = fmaxf(rmax, __shfl_xor(rmax, m, 32));
            const float mnew  = fmaxf(mrow[i], rmax);
            const float alpha = __expf(mrow[i] - mnew);
            const float p0 = __expf(s0v[i] - mnew);
            const float p1 = __expf(s1v[i] - mnew);
            float rs = p0 + p1;
#pragma unroll
            for (int m = 1; m < 16; m <<= 1)
                rs += __shfl_xor(rs, m, 32);
            lrow[i] = lrow[i] * alpha + rs;
            mrow[i] = mnew;
#pragma unroll
            for (int j = 0; j < 4; ++j) oc[j][i] *= alpha;
            const int row = i + 8 * hi;
            pbuf[wave][row][lml]      = (half_t)p0;
            pbuf[wave][row][16 + lml] = (half_t)p1;
        }

        // C-layout -> A-layout through LDS (same wave; LDS in-order)
        asm volatile("s_wait_dscnt 0x0" ::: "memory");
        v16h pa;
#pragma unroll
        for (int v = 0; v < 8; ++v) {
            const int kk = a_kbase(v, hi);
            pa[2 * v]     = pbuf[wave][lml][kk];
            pa[2 * v + 1] = pbuf[wave][lml][kk + 1];
        }
        asm volatile("s_wait_dscnt 0x0" ::: "memory");

        // ---- O += P V (V tile from LDS, key-pairs contiguous) ----
#pragma unroll
        for (int j = 0; j < 4; ++j) {
            v16h bv;
#pragma unroll
            for (int v = 0; v < 8; ++v) {
                const int kp = a_kbase(v, hi);
                bv[2 * v]     = vTile[cur][j * 16 + lml][kp];
                bv[2 * v + 1] = vTile[cur][j * 16 + lml][kp + 1];
            }
            oc[j] = wmma_f16(pa, bv, oc[j]);
        }

        // All waves done reading buf[cur] before it is refilled in it+2.
        __syncthreads();
    }

    // ---- normalize and emit f16 attention output [b][q][h*64+d] ----
#pragma unroll
    for (int j = 0; j < 4; ++j)
#pragma unroll
        for (int i = 0; i < 8; ++i) {
            const int row = i + 8 * hi;
            const int q   = qb0 + row;
            const float o = oc[j][i] / lrow[i];
            Ob[((long)b * SEQ + q) * HIDDEN + h * HDIM + j * 16 + lml] = (half_t)o;
        }
}

extern "C" void kernel_launch(void* const* d_in, const int* in_sizes, int n_in,
                              void* d_out, int out_size, void* d_ws, size_t ws_size,
                              hipStream_t stream) {
    const float* qh    = (const float*)d_in[0];
    const float* kh    = (const float*)d_in[1];
    const float* amask = (const float*)d_in[2];
    const int*   align = (const int*)d_in[3];
    const float* Wq    = (const float*)d_in[4];
    const float* Wk    = (const float*)d_in[5];
    const float* Wv    = (const float*)d_in[6];
    const float* Wo    = (const float*)d_in[7];

    char* ws = (char*)d_ws;
    half_t* Qb = (half_t*)(ws);                              // 4 MB
    half_t* Kb = (half_t*)(ws + (size_t)4  * 1024 * 1024);   // 4 MB
    half_t* Vt = (half_t*)(ws + (size_t)8  * 1024 * 1024);   // 4 MB
    half_t* Ab = (half_t*)(ws + (size_t)12 * 1024 * 1024);   // 4 MB

    const int M     = BATCH * SEQ;               // 8192 rows
    const int tiles = (M / 16) * (HIDDEN / 16);  // 8192 wave tiles
    dim3 blk(32, 4);
    dim3 g1(tiles / 4);

    proj_kernel<0><<<g1, blk, 0, stream>>>(qh, Wq, Qb);
    proj_kernel<1><<<g1, blk, 0, stream>>>(kh, Wk, Kb);
    proj_kernel<2><<<g1, blk, 0, stream>>>(kh, Wv, Vt);
    attn_kernel<<<dim3(SEQ / 16 / 4, NHEADS, BATCH), blk, 0, stream>>>(
        Qb, Kb, Vt, amask, align, Ab);
    proj_kernel<3><<<g1, blk, 0, stream>>>(Ab, Wo, (float*)d_out);
}
